// Exophormer_GNN_18674517803416
// MI455X (gfx1250) — compile-verified, matrix operand order checked by reading, not run on hardware
//
#include <hip/hip_runtime.h>

typedef __attribute__((ext_vector_type(16))) _Float16 v16h;
typedef __attribute__((ext_vector_type(8)))  float    v8f;

#define HID     128
#define HEADS   8
#define CH      16
#define N_REAL  20000
#define N_GRAPHS 20
#define VIRT    4
#define N_TOTAL (N_REAL + N_GRAPHS * VIRT)      /* 20080 */
#define N_E0    640000
#define VE      (VIRT * N_TOTAL)                /* 80320 */
#define E_TOTAL (N_E0 + N_REAL + VE)            /* 740320 */
#define NB_SCAN ((N_TOTAL + 255) / 256)         /* 79 */

// ---------------------------------------------------------------------------
// Weight prep: f32 row-major [K=128][N=128] -> f16 in CDNA5 B-fragment layout
// dst layout: [tile(8)][kstep(4)][lane(32)][j(16)] ; per (lane,j):
//   n = 16*tile + (lane&15),  k = 32*kstep + klocal(lane>>4, j)
//   klocal = (j<8) ? 8*hi + j : 16 + 8*hi + (j-8)     (mirrors 16-bit A layout)
// ---------------------------------------------------------------------------
__global__ __launch_bounds__(256) void k_prep_weight(const float* __restrict__ W,
                                                     _Float16* __restrict__ dst) {
  int t = blockIdx.x * 256 + threadIdx.x;
  if (t >= HID * HID) return;
  int j    = t & 15;
  int lane = (t >> 4) & 31;
  int s    = (t >> 9) & 3;
  int tile = t >> 11;
  int hi = lane >> 4;
  int kl = (j < 8) ? (8 * hi + j) : (16 + 8 * hi + (j - 8));
  int kk = 32 * s + kl;
  int n  = 16 * tile + (lane & 15);
  dst[t] = (_Float16)W[kk * HID + n];
}

// ---------------------------------------------------------------------------
// Fused Q/K/V/skip GEMM: one block = 16 rows of h; 8 waves = 8 col tiles.
// A fragments loaded once (f32->f16), reused for all 4 weight matrices.
// 16 x v_wmma_f32_16x16x32_f16 per wave.
// ---------------------------------------------------------------------------
struct GemmArgs {
  const float* h;
  const _Float16* w;               // [mat(4)][tile(8)][kstep(4)][lane(32)][j(16)]
  const float *bq, *bk, *bv, *bs;
  float *q, *k, *v, *hn;
};

__global__ __launch_bounds__(256) void k_gemm_qkvs(GemmArgs A) {
  const int lane = threadIdx.x & 31;
  const int wv   = threadIdx.x >> 5;           // col tile 0..7
  const int row0 = blockIdx.x * 16;
  const int hi   = lane >> 4;
  const int kb   = 8 * hi;
  const float* hrow = A.h + (size_t)(row0 + (lane & 15)) * HID;

  v16h a[4];
#pragma unroll
  for (int s = 0; s < 4; ++s) {
#pragma unroll
    for (int j = 0; j < 8; ++j) {
      a[s][j]     = (_Float16)hrow[32 * s + kb + j];
      a[s][j + 8] = (_Float16)hrow[32 * s + 16 + kb + j];
    }
  }

  const int col = wv * 16 + (lane & 15);
  float* outp[4]       = {A.q, A.k, A.v, A.hn};
  const float* bias[4] = {A.bq, A.bk, A.bv, A.bs};

#pragma unroll
  for (int m = 0; m < 4; ++m) {
    v8f c = {};
    const _Float16* wb = A.w + ((size_t)m * 8 + wv) * (4 * 32 * 16);
#pragma unroll
    for (int s = 0; s < 4; ++s) {
      v16h b = *(const v16h*)(wb + (s * 32 + lane) * 16);
      c = __builtin_amdgcn_wmma_f32_16x16x32_f16(false, a[s], false, b,
                                                 (short)0, c, false, false);
    }
    float bb = bias[m][col];
    float* op = outp[m] + (size_t)(row0 + 8 * hi) * HID + col;
#pragma unroll
    for (int r = 0; r < 8; ++r) op[(size_t)r * HID] = c[r] + bb;
  }
}

// ---------------------------------------------------------------------------
// Virtual-edge construction (mirrors the reference's torch/np index logic)
// ---------------------------------------------------------------------------
__global__ __launch_bounds__(256) void k_zero_u32(unsigned* p, int n) {
  int i = blockIdx.x * 256 + threadIdx.x;
  if (i < n) p[i] = 0u;
}

__global__ __launch_bounds__(256) void k_count_batch(const int* __restrict__ batch,
                                                     int* __restrict__ cnt) {
  int i = blockIdx.x * 256 + threadIdx.x;
  if (i < N_REAL) atomicAdd(&cnt[batch[i]], 1);
}

__global__ void k_prefix(const int* __restrict__ cnt, int* __restrict__ off) {
  if (threadIdx.x == 0 && blockIdx.x == 0) {
    int a = 0;
    off[0] = 0;
    for (int i = 0; i < N_GRAPHS; ++i) { a += VIRT * (cnt[i] + VIRT); off[i + 1] = a; }
  }
}

__device__ __forceinline__ int virt_node(int j, const int* off) {
  int lo = 0, hi = N_GRAPHS - 1;
  while (lo < hi) {
    int mid = (lo + hi + 1) >> 1;
    if (off[mid] <= j) lo = mid; else hi = mid - 1;
  }
  return N_REAL + VIRT * lo + ((j - off[lo]) & (VIRT - 1));
}

__global__ __launch_bounds__(256) void k_fill_edges(const int* __restrict__ ei0,
                                                    const int* __restrict__ off,
                                                    int* __restrict__ oei) {
  int j = blockIdx.x * 256 + threadIdx.x;
  if (j >= E_TOTAL) return;
  int s, d;
  if (j < N_E0) { s = ei0[j]; d = ei0[N_E0 + j]; }
  else {
    int jj = j - N_E0;
    s = (jj < N_REAL) ? jj : virt_node(jj - N_REAL, off);
    d = (jj < VE) ? virt_node(jj, off) : (jj - VE);
  }
  oei[j] = s;
  oei[E_TOTAL + j] = d;
}

// ---------------------------------------------------------------------------
// CSR-by-dst build (done once per call; reused by all 4 layers)
// ---------------------------------------------------------------------------
__global__ __launch_bounds__(256) void k_count_deg(const int* __restrict__ ei,
                                                   int* __restrict__ deg) {
  int e = blockIdx.x * 256 + threadIdx.x;
  if (e < E_TOTAL) atomicAdd(&deg[ei[E_TOTAL + e]], 1);
}

__global__ __launch_bounds__(256) void k_scan1(const int* __restrict__ deg,
                                               int* __restrict__ incl,
                                               int* __restrict__ bsum) {
  __shared__ int sm[256];
  int i = blockIdx.x * 256 + threadIdx.x;
  sm[threadIdx.x] = (i < N_TOTAL) ? deg[i] : 0;
  __syncthreads();
#pragma unroll
  for (int off = 1; off < 256; off <<= 1) {
    int t = (threadIdx.x >= off) ? sm[threadIdx.x - off] : 0;
    __syncthreads();
    sm[threadIdx.x] += t;
    __syncthreads();
  }
  if (i < N_TOTAL) incl[i] = sm[threadIdx.x];
  if (threadIdx.x == 255) bsum[blockIdx.x] = sm[255];
}

__global__ void k_scan2(const int* __restrict__ bsum, int* __restrict__ boff,
                        int* __restrict__ rowptr) {
  if (threadIdx.x == 0 && blockIdx.x == 0) {
    int a = 0;
    for (int i = 0; i < NB_SCAN; ++i) { boff[i] = a; a += bsum[i]; }
    rowptr[N_TOTAL] = a;
  }
}

__global__ __launch_bounds__(256) void k_scan3(const int* __restrict__ incl,
                                               const int* __restrict__ deg,
                                               const int* __restrict__ boff,
                                               int* __restrict__ rowptr,
                                               int* __restrict__ cursor) {
  int i = blockIdx.x * 256 + threadIdx.x;
  if (i >= N_TOTAL) return;
  int ex = incl[i] - deg[i] + boff[blockIdx.x];
  rowptr[i] = ex;
  cursor[i] = ex;
}

__global__ __launch_bounds__(256) void k_scatter(const int* __restrict__ ei,
                                                 int* __restrict__ cursor,
                                                 int* __restrict__ csr_src,
                                                 int* __restrict__ csr_eid) {
  int e = blockIdx.x * 256 + threadIdx.x;
  if (e >= E_TOTAL) return;
  int d = ei[E_TOTAL + e];
  int p = atomicAdd(&cursor[d], 1);
  csr_src[p] = ei[e];
  csr_eid[p] = e;
}

// ---------------------------------------------------------------------------
// h0 init and output copy
// ---------------------------------------------------------------------------
__global__ __launch_bounds__(256) void k_init_h(const float* __restrict__ x,
                                                const float* __restrict__ vemb,
                                                float* __restrict__ h) {
  int i = blockIdx.x * 256 + threadIdx.x;
  if (i >= N_TOTAL * HID) return;
  int node = i >> 7;
  h[i] = (node < N_REAL) ? x[i] : vemb[((node - N_REAL) & 3) * HID + (i & 127)];
}

__global__ __launch_bounds__(256) void k_copy_out(const float* __restrict__ h,
                                                  float* __restrict__ out) {
  int i = blockIdx.x * 256 + threadIdx.x;
  if (i < N_REAL * HID) out[i] = h[i];
}

// ---------------------------------------------------------------------------
// Fused attention: block = node, wave32 = head. q in registers, lanes stride
// the node's incoming edges; softmax via wave shfl butterflies; alpha*v
// partials reduced through LDS; h_next (preloaded with skip) updated with
// plain stores — no global atomics in the per-layer hot path.
// ---------------------------------------------------------------------------
__global__ __launch_bounds__(256) void k_attn(const float* __restrict__ q,
                                              const float* __restrict__ k,
                                              const float* __restrict__ v,
                                              const int* __restrict__ rowptr,
                                              const int* __restrict__ csr_src,
                                              const int* __restrict__ csr_eid,
                                              float* __restrict__ Lb,
                                              float* __restrict__ alpha_out,
                                              int write_alpha,
                                              float* __restrict__ hn) {
  __shared__ float red[HEADS][32][CH];
  const int node = blockIdx.x;
  const int head = threadIdx.x >> 5;
  const int lane = threadIdx.x & 31;
  const int r0 = rowptr[node];
  const int r1 = rowptr[node + 1];

  const float4* qp = (const float4*)(q + (size_t)node * HID + head * CH);
  const float4 q0 = qp[0], q1 = qp[1], q2 = qp[2], q3 = qp[3];

  // pass 1: logits + wave max
  float m = -__FLT_MAX__;
  for (int p = r0 + lane; p < r1; p += 32) {
    int s = csr_src[p];
    const float4* kp = (const float4*)(k + (size_t)s * HID + head * CH);
    float4 k0 = kp[0], k1 = kp[1], k2 = kp[2], k3 = kp[3];
    float acc = q0.x * k0.x + q0.y * k0.y + q0.z * k0.z + q0.w * k0.w
              + q1.x * k1.x + q1.y * k1.y + q1.z * k1.z + q1.w * k1.w
              + q2.x * k2.x + q2.y * k2.y + q2.z * k2.z + q2.w * k2.w
              + q3.x * k3.x + q3.y * k3.y + q3.z * k3.z + q3.w * k3.w;
    float lg = acc * 0.25f;                     // 1/sqrt(16)
    Lb[(size_t)p * HEADS + head] = lg;
    m = fmaxf(m, lg);
  }
#pragma unroll
  for (int off = 16; off; off >>= 1) m = fmaxf(m, __shfl_xor(m, off));

  // pass 2: exp + wave sum (exp stored back into Lb)
  float ssum = 0.f;
  for (int p = r0 + lane; p < r1; p += 32) {
    size_t ix = (size_t)p * HEADS + head;
    float ev = __expf(Lb[ix] - m);
    Lb[ix] = ev;
    ssum += ev;
  }
#pragma unroll
  for (int off = 16; off; off >>= 1) ssum += __shfl_xor(ssum, off);
  float inv = (ssum > 0.f) ? (1.f / ssum) : 0.f;

  // pass 3: alpha, optional alpha output, and alpha * v accumulation
  float4 a0 = {0, 0, 0, 0}, a1 = {0, 0, 0, 0}, a2 = {0, 0, 0, 0}, a3 = {0, 0, 0, 0};
  for (int p = r0 + lane; p < r1; p += 32) {
    float a = Lb[(size_t)p * HEADS + head] * inv;
    if (write_alpha) alpha_out[(size_t)csr_eid[p] * HEADS + head] = a;
    int s = csr_src[p];
    const float4* vp = (const float4*)(v + (size_t)s * HID + head * CH);
    float4 v0 = vp[0], v1 = vp[1], v2 = vp[2], v3 = vp[3];
    a0.x += a * v0.x; a0.y += a * v0.y; a0.z += a * v0.z; a0.w += a * v0.w;
    a1.x += a * v1.x; a1.y += a * v1.y; a1.z += a * v1.z; a1.w += a * v1.w;
    a2.x += a * v2.x; a2.y += a * v2.y; a2.z += a * v2.z; a2.w += a * v2.w;
    a3.x += a * v3.x; a3.y += a * v3.y; a3.z += a * v3.z; a3.w += a * v3.w;
  }

  // cross-lane channel reduction through LDS (trip counts uniform per block)
  float* rp = &red[head][lane][0];
  rp[0] = a0.x;  rp[1] = a0.y;  rp[2] = a0.z;  rp[3] = a0.w;
  rp[4] = a1.x;  rp[5] = a1.y;  rp[6] = a1.z;  rp[7] = a1.w;
  rp[8] = a2.x;  rp[9] = a2.y;  rp[10] = a2.z; rp[11] = a2.w;
  rp[12] = a3.x; rp[13] = a3.y; rp[14] = a3.z; rp[15] = a3.w;
  __syncthreads();
  if (lane < CH) {
    float t = 0.f;
    for (int i = 0; i < 32; ++i) t += red[head][i][lane];
    size_t oix = (size_t)node * HID + head * CH + lane;
    hn[oix] += t;                                // hn preloaded with h@Ws + bs
  }
}

// ---------------------------------------------------------------------------
// Host driver
// ---------------------------------------------------------------------------
extern "C" void kernel_launch(void* const* d_in, const int* in_sizes, int n_in,
                              void* d_out, int out_size, void* d_ws, size_t ws_size,
                              hipStream_t stream) {
  (void)out_size; (void)ws_size; (void)n_in;
  const float* x     = (const float*)d_in[0];
  const int*   ei0   = (const int*)d_in[1];
  const int*   batch = (const int*)d_in[2];

  // Params: 4 layers x {Wq,bq,Wk,bk,Wv,bv,Ws,bs}, then virt_emb.
  // Detect insertion order (W,b alternating) vs sorted pytree order via sizes.
  const float* W[4][4];
  const float* B[4][4];
  bool alt = (in_sizes[4] == HID);
  for (int l = 0; l < 4; ++l) {
    int base = 3 + l * 8;
    if (alt) {
      W[l][0] = (const float*)d_in[base + 0]; B[l][0] = (const float*)d_in[base + 1];
      W[l][1] = (const float*)d_in[base + 2]; B[l][1] = (const float*)d_in[base + 3];
      W[l][2] = (const float*)d_in[base + 4]; B[l][2] = (const float*)d_in[base + 5];
      W[l][3] = (const float*)d_in[base + 6]; B[l][3] = (const float*)d_in[base + 7];
    } else {  // sorted: Wk,Wq,Ws,Wv,bk,bq,bs,bv
      W[l][1] = (const float*)d_in[base + 0];
      W[l][0] = (const float*)d_in[base + 1];
      W[l][3] = (const float*)d_in[base + 2];
      W[l][2] = (const float*)d_in[base + 3];
      B[l][1] = (const float*)d_in[base + 4];
      B[l][0] = (const float*)d_in[base + 5];
      B[l][3] = (const float*)d_in[base + 6];
      B[l][2] = (const float*)d_in[base + 7];
    }
  }
  const float* vemb = (const float*)d_in[35];

  // Output sections: h[:20000] f32 | full_ei 2*E int | alpha E*8 f32
  float* out_h     = (float*)d_out;
  int*   out_ei    = (int*)(out_h + (size_t)N_REAL * HID);
  float* out_alpha = (float*)(out_ei + 2 * (size_t)E_TOTAL);

  // Workspace carve
  char* ws = (char*)d_ws;
  size_t o = 0;
  auto carve = [&](size_t bytes) -> void* {
    void* p = ws + o;
    o = (o + bytes + 255) & ~(size_t)255;
    return p;
  };
  _Float16* wf16 = (_Float16*)carve((size_t)16 * HID * HID * sizeof(_Float16));
  int* cnt    = (int*)carve(N_GRAPHS * sizeof(int));
  int* off    = (int*)carve((N_GRAPHS + 1) * sizeof(int));
  int* deg    = (int*)carve((size_t)N_TOTAL * sizeof(int));
  int* incl   = (int*)carve((size_t)N_TOTAL * sizeof(int));
  int* bsum   = (int*)carve(NB_SCAN * sizeof(int));
  int* boff   = (int*)carve(NB_SCAN * sizeof(int));
  int* rowptr = (int*)carve((size_t)(N_TOTAL + 1) * sizeof(int));
  int* cursor = (int*)carve((size_t)N_TOTAL * sizeof(int));
  int* csr_src = (int*)carve((size_t)E_TOTAL * sizeof(int));
  int* csr_eid = (int*)carve((size_t)E_TOTAL * sizeof(int));
  float* qb = (float*)carve((size_t)N_TOTAL * HID * sizeof(float));
  float* kb = (float*)carve((size_t)N_TOTAL * HID * sizeof(float));
  float* vb = (float*)carve((size_t)N_TOTAL * HID * sizeof(float));
  float* h0 = (float*)carve((size_t)N_TOTAL * HID * sizeof(float));
  float* h1 = (float*)carve((size_t)N_TOTAL * HID * sizeof(float));
  float* Lb = (float*)carve((size_t)E_TOTAL * HEADS * sizeof(float));

  // 1) weights -> f16 WMMA layout
  for (int l = 0; l < 4; ++l)
    for (int m = 0; m < 4; ++m)
      k_prep_weight<<<(HID * HID) / 256, 256, 0, stream>>>(
          W[l][m], wf16 + ((size_t)l * 4 + m) * HID * HID);

  // 2) virtual edges (writes full_ei output, reused as edge list)
  k_zero_u32<<<1, 64, 0, stream>>>((unsigned*)cnt, N_GRAPHS);
  k_count_batch<<<(N_REAL + 255) / 256, 256, 0, stream>>>(batch, cnt);
  k_prefix<<<1, 1, 0, stream>>>(cnt, off);
  k_fill_edges<<<(E_TOTAL + 255) / 256, 256, 0, stream>>>(ei0, off, out_ei);

  // 3) CSR-by-dst (shared by all layers)
  k_zero_u32<<<(N_TOTAL + 255) / 256, 256, 0, stream>>>((unsigned*)deg, N_TOTAL);
  k_count_deg<<<(E_TOTAL + 255) / 256, 256, 0, stream>>>(out_ei, deg);
  k_scan1<<<NB_SCAN, 256, 0, stream>>>(deg, incl, bsum);
  k_scan2<<<1, 1, 0, stream>>>(bsum, boff, rowptr);
  k_scan3<<<NB_SCAN, 256, 0, stream>>>(incl, deg, boff, rowptr, cursor);
  k_scatter<<<(E_TOTAL + 255) / 256, 256, 0, stream>>>(out_ei, cursor, csr_src, csr_eid);

  // 4) h0 = concat(x, virt_emb[virtual_nodes])
  k_init_h<<<(N_TOTAL * HID + 255) / 256, 256, 0, stream>>>(x, vemb, h0);

  // 5) layers
  float* hc = h0;
  float* hn = h1;
  for (int l = 0; l < 4; ++l) {
    GemmArgs ga{hc, wf16 + (size_t)l * 4 * HID * HID,
                B[l][0], B[l][1], B[l][2], B[l][3], qb, kb, vb, hn};
    k_gemm_qkvs<<<N_TOTAL / 16, 256, 0, stream>>>(ga);
    k_attn<<<N_TOTAL, 256, 0, stream>>>(qb, kb, vb, rowptr, csr_src, csr_eid,
                                        Lb, (l == 3) ? out_alpha : Lb,
                                        (l == 3) ? 1 : 0, hn);
    float* t = hc; hc = hn; hn = t;
  }

  // 6) output h[:N_REAL]
  k_copy_out<<<(N_REAL * HID + 255) / 256, 256, 0, stream>>>(hc, out_h);
}
